// Attention_19026705121921
// MI455X (gfx1250) — compile-verified, matrix operand order checked by reading, not run on hardware
//
#include <hip/hip_runtime.h>
#include <hip/hip_bf16.h>
#include <cstdint>
#include <math.h>

typedef __attribute__((ext_vector_type(16))) _Float16 v16h;
typedef __attribute__((ext_vector_type(8)))  _Float16 v8h;
typedef __attribute__((ext_vector_type(8)))  float    v8f;
typedef __attribute__((ext_vector_type(4)))  float    v4f;

#define B_    32
#define N_    480
#define DIM_  1024
#define H_    16
#define D_    64
#define M_TOT (B_ * N_)   // 15360
#define LOG2E 1.4426950408889634f

// ---------------------------------------------------------------------------
// CDNA5 async copy: global -> LDS, 128b per lane, tracked by ASYNCcnt.
// GVS addressing: 64-bit SGPR base + 32-bit per-lane byte offset.
// LDS dest = low 32 bits of the generic shared pointer (flat aperture layout).
// ---------------------------------------------------------------------------
__device__ __forceinline__ void async_b128(const void* lds_dst, uint32_t gbyte,
                                           const void* sbase) {
  asm volatile("global_load_async_to_lds_b128 %0, %1, %2"
               :
               : "v"((uint32_t)(uintptr_t)lds_dst), "v"(gbyte), "s"(sbase)
               : "memory");
}
__device__ __forceinline__ void wait_async0() {
  asm volatile("s_wait_asynccnt 0x0" ::: "memory");
}

// ---------------------------------------------------------------------------
// 16-lane xor-butterfly max via v_permlane16_b32 (pure VALU, no DS round trip).
// Selector nibbles: lane i (0..7) from S1, lane i (8..15) from S2.
// ---------------------------------------------------------------------------
__device__ __forceinline__ float pl16_max(float x, unsigned selLo, unsigned selHi) {
  const int xi = __float_as_int(x);
  const int yi = __builtin_amdgcn_permlane16(xi, xi, selLo, selHi, false, false);
  return fmaxf(x, __int_as_float(yi));
}
__device__ __forceinline__ float red_max16(float x) {
  x = pl16_max(x, 0x67452301u, 0xEFCDAB89u);  // xor 1
  x = pl16_max(x, 0x54761032u, 0xDCFE98BAu);  // xor 2
  x = pl16_max(x, 0x32107654u, 0xBA98FEDCu);  // xor 4
  x = pl16_max(x, 0xFEDCBA98u, 0x76543210u);  // xor 8
  return x;
}

// Raw hardware 2^x (v_exp_f32). Softmax args are <= 0; HW flush below the
// normal range is exactly the desired behavior — skip the libm range fixup.
__device__ __forceinline__ float fast_exp2(float x) {
  return __builtin_amdgcn_exp2f(x);
}

// ---------------------------------------------------------------------------
// Fragment loader: 16-bit A/B matrix layout for V_WMMA_*_16X16X32 (wave32).
// lane L: row/col index = L&15 ; halves 0..7  = K[hi*8 .. hi*8+7]
//                                halves 8..15 = K[16+hi*8 .. 16+hi*8+7]
// ---------------------------------------------------------------------------
__device__ __forceinline__ v16h load_frag(const _Float16* base, int ldm, int lane) {
  const int g  = lane & 15;
  const int hi = (lane >> 4) & 1;
  const _Float16* p = base + g * ldm + hi * 8;
  v8h lo  = *(const v8h*)(p);
  v8h hi8 = *(const v8h*)(p + 16);
  v16h r;
#pragma unroll
  for (int i = 0; i < 8; ++i) { r[i] = lo[i]; r[i + 8] = hi8[i]; }
  return r;
}

__device__ __forceinline__ v8f wmma_f16(v16h a, v16h b, v8f c) {
  return __builtin_amdgcn_wmma_f32_16x16x32_f16(false, a, false, b, (short)0, c,
                                                false, false);
}

// ---------------------------------------------------------------------------
// fp32 -> f16 conversion (grid-stride)
// ---------------------------------------------------------------------------
__global__ void cvt_kernel(const float* __restrict__ s, _Float16* __restrict__ d, int n) {
  int i = blockIdx.x * blockDim.x + threadIdx.x;
  int stride = gridDim.x * blockDim.x;
  for (; i < n; i += stride) d[i] = (_Float16)s[i];
}

// ---------------------------------------------------------------------------
// pe (480x480 fp32) -> peT[key][qrow] * LOG2E  (base-2 softmax domain), so the
// attention bias fetch is two contiguous b128 loads per half-group.
// ---------------------------------------------------------------------------
__global__ __launch_bounds__(256) void transpose_pe(const float* __restrict__ pe,
                                                    float* __restrict__ peT) {
  __shared__ float tile[32][33];
  const int bx = blockIdx.x * 32, by = blockIdx.y * 32;
  const int tx = threadIdx.x & 31, ty = threadIdx.x >> 5;  // 32 x 8
#pragma unroll
  for (int r = ty; r < 32; r += 8) tile[r][tx] = pe[(by + r) * N_ + bx + tx];
  __syncthreads();
#pragma unroll
  for (int r = ty; r < 32; r += 8)
    peT[(bx + r) * N_ + by + tx] = tile[tx][r] * LOG2E;
}

// ---------------------------------------------------------------------------
// Tiled WMMA GEMM:  C[M,N] = A[M,K] * W[N,K]^T   (M=15360, N=K=1024)
// Block tile 128x128, 8 waves (wave tile 32x64 = 2x4 accumulators), BK=64.
// Ping-pong LDS buffers filled by GLOBAL_LOAD_ASYNC_TO_LDS_B128; one barrier
// per K-step; the async DMA of tile i+1 overlaps the 16 WMMAs of tile i.
// ---------------------------------------------------------------------------
template <bool HEAD_SPLIT>
__global__ __launch_bounds__(256) void gemm_kernel(
    const _Float16* __restrict__ A, const _Float16* __restrict__ W,
    const float* __restrict__ bias, _Float16* __restrict__ outH,
    float* __restrict__ outF) {
  constexpr int BM = 128, BN = 128, BK = 64, LDT = 72;
  constexpr int NITER = DIM_ / BK;  // 16
  __shared__ alignas(16) _Float16 As[2][BM][LDT];
  __shared__ alignas(16) _Float16 Bs[2][BN][LDT];

  const int t    = threadIdx.x;
  const int lane = t & 31;
  const int wid  = t >> 5;
  const int m0   = (wid & 3) * 32;
  const int n0   = (wid >> 2) * 64;
  const int blockM = blockIdx.y * BM;
  const int blockN = blockIdx.x * BN;

  v8f acc[2][4] = {};

  auto issue = [&](int buf, int k0) {
#pragma unroll
    for (int u = 0; u < 4; ++u) {           // 4 chunks of 8 halves per matrix
      const int c   = t * 4 + u;            // 1024 chunks over 256 threads
      const int row = c >> 3;
      const int off = (c & 7) * 8;
      const uint32_t ga = (uint32_t)(((blockM + row) * DIM_ + k0 + off) * 2);
      const uint32_t gb = (uint32_t)(((blockN + row) * DIM_ + k0 + off) * 2);
      async_b128(&As[buf][row][off], ga, A);
      async_b128(&Bs[buf][row][off], gb, W);
    }
  };

  issue(0, 0);
  for (int it = 0; it < NITER; ++it) {
    wait_async0();
    __syncthreads();
    if (it + 1 < NITER) issue((it + 1) & 1, (it + 1) * BK);
    const int buf = it & 1;
#pragma unroll
    for (int kc = 0; kc < 2; ++kc) {
      v16h af[2], bf[4];
#pragma unroll
      for (int i = 0; i < 2; ++i)
        af[i] = load_frag(&As[buf][m0 + i * 16][kc * 32], LDT, lane);
#pragma unroll
      for (int j = 0; j < 4; ++j)
        bf[j] = load_frag(&Bs[buf][n0 + j * 16][kc * 32], LDT, lane);
#pragma unroll
      for (int i = 0; i < 2; ++i)
#pragma unroll
        for (int j = 0; j < 4; ++j) acc[i][j] = wmma_f16(af[i], bf[j], acc[i][j]);
    }
  }

  const int g   = lane & 15;
  const int hiL = (lane >> 4) & 1;
#pragma unroll
  for (int i = 0; i < 2; ++i) {
#pragma unroll
    for (int j = 0; j < 4; ++j) {
      const int n = blockN + n0 + j * 16 + g;
      float bv = 0.f;
      if constexpr (!HEAD_SPLIT) bv = bias[n];
#pragma unroll
      for (int r = 0; r < 8; ++r) {
        const int m   = blockM + m0 + i * 16 + r + hiL * 8;
        const float v = acc[i][j][r];
        if constexpr (HEAD_SPLIT) {
          const int b   = m / N_;
          const int row = m - b * N_;
          const int h   = n >> 6;
          const int dd  = n & 63;
          outH[(((size_t)(b * H_ + h)) * N_ + row) * D_ + dd] = (_Float16)v;
        } else {
          outF[(size_t)m * DIM_ + n] = v + bv;
        }
      }
    }
  }
}

// ---------------------------------------------------------------------------
// Flash attention: one block = one (b,h), 160 query rows (10 waves x 16 rows).
// K tile via async-to-LDS; pe bias from peT (pre-scaled by log2 e); row sums
// accumulated by an all-ones WMMA column (o[4]); row max via v_permlane16.
// ---------------------------------------------------------------------------
__global__ __launch_bounds__(320) void attn_kernel(
    const _Float16* __restrict__ Qp, const _Float16* __restrict__ Kp,
    const _Float16* __restrict__ Vp, const float* __restrict__ peT,
    _Float16* __restrict__ outA) {
  constexpr int TK = 32;
  __shared__ alignas(16) _Float16 Ks[TK][72];      // key-row major (B-frag layout for QK^T)
  __shared__ alignas(16) _Float16 Vt[D_][40];      // transposed V (B-frag layout for PV)
  __shared__ alignas(16) _Float16 Ps[10][16][40];  // per-wave P staging

  const int t     = threadIdx.x;
  const int lane  = t & 31;
  const int wid   = t >> 5;  // 0..9
  const int bh    = blockIdx.x;
  const int qbase = blockIdx.y * 160 + wid * 16;
  const int g     = lane & 15;
  const int hiL   = (lane >> 4) & 1;
  const int b     = bh >> 4;
  const int h     = bh & 15;

  const _Float16* Qrow = Qp + ((size_t)bh * N_ + qbase) * D_;
  const v16h qa0 = load_frag(Qrow, D_, lane);
  const v16h qa1 = load_frag(Qrow + 32, D_, lane);

  v16h ones;
#pragma unroll
  for (int i = 0; i < 16; ++i) ones[i] = (_Float16)1.0f;

  const float scale = 0.125f * LOG2E;  // 1/sqrt(64), base-2 softmax domain
  float mrow[8];
  v8f o[5] = {};  // o[0..3]: output d-tiles, o[4]: running row sums (l)
#pragma unroll
  for (int r = 0; r < 8; ++r) mrow[r] = -1e30f;

  for (int kb = 0; kb < N_; kb += TK) {
    __syncthreads();
    if (t < 256) {
      const int row = t >> 3;
      const int off = (t & 7) * 8;
      async_b128(&Ks[row][off],
                 (uint32_t)((((size_t)bh * N_ + kb + row) * D_ + off) * 2), Kp);
      const v8h vv = *(const v8h*)&Vp[((size_t)bh * N_ + kb + row) * D_ + off];
#pragma unroll
      for (int u = 0; u < 8; ++u) Vt[off + u][row] = vv[u];
    }
    wait_async0();
    __syncthreads();

    // S = Q K^T (16 x 32 per wave)
    v8f s[2];
#pragma unroll
    for (int nt = 0; nt < 2; ++nt) {
      v8f a = {};
      const v16h b0 = load_frag(&Ks[nt * 16][0], 72, lane);
      const v16h b1 = load_frag(&Ks[nt * 16][32], 72, lane);
      a = wmma_f16(qa0, b0, a);
      a = wmma_f16(qa1, b1, a);
      s[nt] = a;
    }

    // scale + positional bias (contiguous column fetch from peT)
#pragma unroll
    for (int nt = 0; nt < 2; ++nt) {
      const int key = kb + nt * 16 + g;
      const v4f* pp = (const v4f*)&peT[(size_t)key * N_ + qbase + hiL * 8];
      const v4f p0 = pp[0], p1 = pp[1];
#pragma unroll
      for (int r = 0; r < 4; ++r) {
        s[nt][r]     = s[nt][r] * scale + p0[r];
        s[nt][r + 4] = s[nt][r + 4] * scale + p1[r];
      }
    }

    // online softmax: row max via permlane butterflies (16-lane half-groups)
    float corr[8];
#pragma unroll
    for (int r = 0; r < 8; ++r) {
      const float tm = red_max16(fmaxf(s[0][r], s[1][r]));
      const float mn = fmaxf(mrow[r], tm);
      corr[r] = fast_exp2(mrow[r] - mn);
      mrow[r] = mn;
    }

    // P = 2^(S - m), staged to per-wave LDS for re-fragmentation as A operand
#pragma unroll
    for (int nt = 0; nt < 2; ++nt)
#pragma unroll
      for (int r = 0; r < 8; ++r)
        Ps[wid][r + hiL * 8][nt * 16 + g] = (_Float16)fast_exp2(s[nt][r] - mrow[r]);

    // rescale running accumulators (o[4] carries l: l = l*corr + rowsum(P))
#pragma unroll
    for (int j = 0; j < 5; ++j)
#pragma unroll
      for (int r = 0; r < 8; ++r) o[j][r] *= corr[r];

    // O += P V ; l += P * ones  (same-wave DS ops are in order)
    const v16h pa = load_frag(&Ps[wid][0][0], 40, lane);
#pragma unroll
    for (int j = 0; j < 4; ++j) {
      const v16h bf = load_frag(&Vt[j * 16][0], 40, lane);
      o[j] = wmma_f16(pa, bf, o[j]);
    }
    o[4] = wmma_f16(pa, ones, o[4]);
  }

  // normalize and write f16 [B, N, DIM] (heads concatenated)
#pragma unroll
  for (int r = 0; r < 8; ++r) {
    const float inv = 1.0f / o[4][r];
    const int row = qbase + r + hiL * 8;
#pragma unroll
    for (int j = 0; j < 4; ++j) {
      const int dd = j * 16 + g;
      outA[((size_t)b * N_ + row) * DIM_ + h * D_ + dd] = (_Float16)(o[j][r] * inv);
    }
  }
}

// ---------------------------------------------------------------------------
// Host-side launch
// ---------------------------------------------------------------------------
extern "C" void kernel_launch(void* const* d_in, const int* in_sizes, int n_in,
                              void* d_out, int out_size, void* d_ws, size_t ws_size,
                              hipStream_t stream) {
  const float* q  = (const float*)d_in[0];
  const float* Wq = (const float*)d_in[1];
  const float* Wk = (const float*)d_in[2];
  const float* Wv = (const float*)d_in[3];
  const float* pe = (const float*)d_in[4];
  const float* Wo = (const float*)d_in[5];
  const float* bo = (const float*)d_in[6];
  float* out = (float*)d_out;

  char* ws = (char*)d_ws;
  size_t off = 0;
  const size_t xBytes = (size_t)M_TOT * DIM_ * sizeof(_Float16);   // 31.5 MB
  const size_t wBytes = (size_t)DIM_ * DIM_ * sizeof(_Float16);    // 2 MB
  _Float16* Xh  = (_Float16*)(ws + off); off += xBytes;
  _Float16* Wqh = (_Float16*)(ws + off); off += wBytes;
  _Float16* Wkh = (_Float16*)(ws + off); off += wBytes;
  _Float16* Wvh = (_Float16*)(ws + off); off += wBytes;
  _Float16* Woh = (_Float16*)(ws + off); off += wBytes;
  _Float16* Qp  = (_Float16*)(ws + off); off += xBytes;
  _Float16* Kp  = (_Float16*)(ws + off); off += xBytes;
  _Float16* Vp  = (_Float16*)(ws + off); off += xBytes;
  _Float16* Ao  = (_Float16*)(ws + off); off += xBytes;
  float*    peT = (float*)(ws + off);    off += (size_t)N_ * N_ * sizeof(float);

  cvt_kernel<<<2048, 256, 0, stream>>>(q, Xh, M_TOT * DIM_);
  cvt_kernel<<<512, 256, 0, stream>>>(Wq, Wqh, DIM_ * DIM_);
  cvt_kernel<<<512, 256, 0, stream>>>(Wk, Wkh, DIM_ * DIM_);
  cvt_kernel<<<512, 256, 0, stream>>>(Wv, Wvh, DIM_ * DIM_);
  cvt_kernel<<<512, 256, 0, stream>>>(Wo, Woh, DIM_ * DIM_);
  transpose_pe<<<dim3(N_ / 32, N_ / 32), 256, 0, stream>>>(pe, peT);

  dim3 gg(DIM_ / 128, M_TOT / 128);  // (8, 120)
  gemm_kernel<true><<<gg, 256, 0, stream>>>(Xh, Wqh, nullptr, Qp, nullptr);
  gemm_kernel<true><<<gg, 256, 0, stream>>>(Xh, Wkh, nullptr, Kp, nullptr);
  gemm_kernel<true><<<gg, 256, 0, stream>>>(Xh, Wvh, nullptr, Vp, nullptr);

  attn_kernel<<<dim3(B_ * H_, N_ / 160), 320, 0, stream>>>(Qp, Kp, Vp, peT, Ao);

  gemm_kernel<false><<<gg, 256, 0, stream>>>(Ao, Woh, bo, nullptr, out);
}
